// MoeLayer_90701119357095
// MI455X (gfx1250) — matrix operand order repaired and checked by reading, not compiled, not run on hardware
//
#include <hip/hip_runtime.h>
#include <hip/hip_bf16.h>
#include <stdint.h>

// ---------------------------------------------------------------------------
// Problem dimensions (fixed by the reference)
// ---------------------------------------------------------------------------
constexpr int Dd  = 1024;        // model dim
constexpr int Ee  = 8;           // experts
constexpr int Ff  = 4096;        // expert hidden dim
constexpr int Tt  = 4 * 2048;    // tokens = B*S = 8192
constexpr int CAPc = 1280;       // expert capacity

// ---------------------------------------------------------------------------
// WMMA types (CDNA5 / gfx1250, wave32)
// ---------------------------------------------------------------------------
typedef __bf16 v16bf __attribute__((ext_vector_type(16)));
typedef float  v8f   __attribute__((ext_vector_type(8)));

union AFrag { uint4 q[2]; v16bf v; };

__device__ __forceinline__ unsigned short f2bf_raw(float f) {
  union { float f; uint32_t u; } c; c.f = f;
  uint32_t u = c.u;
  u += 0x7FFFu + ((u >> 16) & 1u);   // round-to-nearest-even bf16
  return (unsigned short)(u >> 16);
}

// CDNA5 async global->LDS copy (ASYNCcnt-tracked); 16 bytes per lane.
__device__ __forceinline__ void async_ld_b128(unsigned lds_off, const void* gptr) {
  unsigned long long ga = (unsigned long long)(size_t)gptr;
  asm volatile("global_load_async_to_lds_b128 %0, %1, off"
               :: "v"(lds_off), "v"(ga) : "memory");
}
__device__ __forceinline__ void wait_async0() {
  asm volatile("s_wait_asynccnt 0x0" ::: "memory");
}

// ---------------------------------------------------------------------------
// 1) init: zero output (capacity-dropped tokens emit 0), slot_token = -1
// ---------------------------------------------------------------------------
__global__ __launch_bounds__(256) void moe_init_kernel(float* __restrict__ out,
                                                       int* __restrict__ slot) {
  size_t i = (size_t)blockIdx.x * 256 + threadIdx.x;
  if (i < (size_t)Tt * Dd) out[i] = 0.0f;
  if (i < (size_t)Ee * CAPc) slot[i] = -1;
}

// ---------------------------------------------------------------------------
// 2) gate: one wave32 per token; dot(1024) x 8 experts; softmax + argmax
// ---------------------------------------------------------------------------
__global__ __launch_bounds__(256) void moe_gate_kernel(
    const float* __restrict__ tokens,   // [T, D]
    const float* __restrict__ gw,       // [D, E]
    int* __restrict__ best_e, float* __restrict__ best_p) {
  const int lane = threadIdx.x & 31;
  const int t = blockIdx.x * 8 + (threadIdx.x >> 5);

  float acc[Ee];
  #pragma unroll
  for (int e = 0; e < Ee; ++e) acc[e] = 0.0f;

  const float* row = tokens + (size_t)t * Dd;
  for (int d = lane; d < Dd; d += 32) {
    const float x = row[d];
    const float* g = gw + (size_t)d * Ee;
    #pragma unroll
    for (int e = 0; e < Ee; ++e) acc[e] += x * g[e];
  }
  #pragma unroll
  for (int e = 0; e < Ee; ++e)
    #pragma unroll
    for (int off = 16; off > 0; off >>= 1)
      acc[e] += __shfl_xor(acc[e], off, 32);

  if (lane == 0) {
    float mx = acc[0]; int ai = 0;
    #pragma unroll
    for (int e = 1; e < Ee; ++e)
      if (acc[e] > mx) { mx = acc[e]; ai = e; }
    float s = 0.0f;
    #pragma unroll
    for (int e = 0; e < Ee; ++e) s += __expf(acc[e] - mx);
    best_e[t] = ai;
    best_p[t] = 1.0f / s;   // exp(mx-mx)/sum == softmax prob of argmax
  }
}

// ---------------------------------------------------------------------------
// 3) scan: deterministic per-expert prefix in token order (single block)
// ---------------------------------------------------------------------------
__global__ __launch_bounds__(256) void moe_scan_kernel(
    const int* __restrict__ best_e,
    int* __restrict__ pos, int* __restrict__ slot) {
  __shared__ int cnt[256][Ee];
  const int tid = threadIdx.x;
  const int t0 = tid * (Tt / 256);

  #pragma unroll
  for (int e = 0; e < Ee; ++e) cnt[tid][e] = 0;
  for (int i = 0; i < Tt / 256; ++i) cnt[tid][best_e[t0 + i]]++;
  __syncthreads();

  if (tid < Ee) {            // serial exclusive scan per expert (256 steps)
    int run = 0;
    for (int i = 0; i < 256; ++i) {
      const int v = cnt[i][tid];
      cnt[i][tid] = run;
      run += v;
    }
  }
  __syncthreads();

  int c[Ee];
  #pragma unroll
  for (int e = 0; e < Ee; ++e) c[e] = cnt[tid][e];
  for (int i = 0; i < Tt / 256; ++i) {
    const int t = t0 + i;
    const int e = best_e[t];
    const int p = c[e]++;
    if (p < CAPc) {
      pos[t] = p;
      slot[e * CAPc + p] = t;
    } else {
      pos[t] = -1;
    }
  }
}

// ---------------------------------------------------------------------------
// 4) dispatch: gather kept tokens into bf16 Xe[E, CAP, D]
// ---------------------------------------------------------------------------
__global__ __launch_bounds__(256) void moe_dispatch_kernel(
    const float* __restrict__ tokens,
    const int* __restrict__ best_e, const int* __restrict__ pos,
    unsigned short* __restrict__ Xe) {
  const int t = blockIdx.x;
  const int p = pos[t];
  if (p < 0) return;
  const int e = best_e[t];
  unsigned short* dst = Xe + ((size_t)e * CAPc + p) * Dd;
  const float* src = tokens + (size_t)t * Dd;
  for (int i = threadIdx.x; i < Dd; i += 256) dst[i] = f2bf_raw(src[i]);
}

// ---------------------------------------------------------------------------
// 5) ffn1: per-expert GEMM  H = relu(Xe @ w1[e]),  M=CAP K=1024 N=4096
//    128x128 block tile, 8 waves x (32x64), v_wmma_f32_16x16x32_bf16,
//    double-buffered LDS; A-tile staged with global_load_async_to_lds_b128.
// ---------------------------------------------------------------------------
__global__ __launch_bounds__(256) void moe_ffn1_kernel(
    const unsigned short* __restrict__ Xe,   // [E, CAP, D] bf16
    const float* __restrict__ w1,            // [E, D, F] fp32
    unsigned short* __restrict__ Hb) {       // [E, CAP, F] bf16
  constexpr int K = Dd, N = Ff, NK = K / 32;
  __shared__ unsigned short As[2][128][32];  // [buf][m][k]
  __shared__ unsigned short Bs[2][128][32];  // [buf][n][k] (transposed stage)

  const int e = blockIdx.z;
  const int mBase = blockIdx.y * 128;
  const int nBase = blockIdx.x * 128;
  const unsigned short* A = Xe + (size_t)e * CAPc * K;
  const float* W = w1 + (size_t)e * K * N;
  unsigned short* Hout = Hb + (size_t)e * CAPc * N;

  const int tid  = threadIdx.x;
  const int lane = tid & 31;
  const int wave = tid >> 5;
  const int m0w  = (wave & 3) * 32;
  const int n0w  = (wave >> 2) * 64;
  const int hlf  = lane >> 4;
  const int lm   = lane & 15;

  v8f acc[2][4];
  const v8f vzero = {0.f,0.f,0.f,0.f,0.f,0.f,0.f,0.f};
  #pragma unroll
  for (int i = 0; i < 2; ++i)
    #pragma unroll
    for (int j = 0; j < 4; ++j) acc[i][j] = vzero;

  const int rA  = tid >> 1, csA = (tid & 1) * 16;   // A stage: 2 thr/row
  const int nB  = tid >> 1, ksB = (tid & 1) * 16;   // B stage: 2 thr/row

  auto stageA = [&](int kb, int buf) {              // async bf16 copy, 32B/thr
    const unsigned dst = (unsigned)(size_t)&As[buf][rA][csA];
    const unsigned short* src = A + (size_t)(mBase + rA) * K + kb * 32 + csA;
    async_ld_b128(dst,      src);
    async_ld_b128(dst + 16, src + 8);
  };
  auto stageB = [&](int kb, int buf) {              // fp32->bf16, [n][k] stage
    const float* src = W + (size_t)(kb * 32 + ksB) * N + (nBase + nB);
    unsigned short tmp[16];
    #pragma unroll
    for (int i = 0; i < 16; ++i) tmp[i] = f2bf_raw(src[(size_t)i * N]);
    uint4* dst = (uint4*)&Bs[buf][nB][ksB];
    dst[0] = *(const uint4*)&tmp[0];
    dst[1] = *(const uint4*)&tmp[8];
  };

  stageA(0, 0);
  stageB(0, 0);

  int buf = 0;
  for (int kb = 0; kb < NK; ++kb) {
    wait_async0();          // this wave's async LDS writes have landed
    __syncthreads();        // tile `buf` published to all waves
    if (kb + 1 < NK) {
      stageA(kb + 1, buf ^ 1);
      stageB(kb + 1, buf ^ 1);
      __builtin_prefetch(W + (size_t)((kb + 1) * 32 + ksB) * N + (nBase + nB), 0, 3);
    }

    AFrag a[2], b[4];
    #pragma unroll
    for (int mt = 0; mt < 2; ++mt) {        // 16-bit A 16x32 fragment layout
      const int m = m0w + mt * 16 + lm;
      a[mt].q[0] = *(const uint4*)&As[buf][m][hlf * 16];
      a[mt].q[1] = *(const uint4*)&As[buf][m][hlf * 16 + 8];
    }
    #pragma unroll
    for (int nt = 0; nt < 4; ++nt) {        // 16-bit B 32x16 fragment layout
      const int n = n0w + nt * 16 + lm;
      b[nt].q[0] = *(const uint4*)&Bs[buf][n][hlf * 16];
      b[nt].q[1] = *(const uint4*)&Bs[buf][n][hlf * 16 + 8];
    }
    #pragma unroll
    for (int mt = 0; mt < 2; ++mt)
      #pragma unroll
      for (int nt = 0; nt < 4; ++nt)
        acc[mt][nt] = __builtin_amdgcn_wmma_f32_16x16x32_bf16(
            false, a[mt].v, false, b[nt].v, (short)0, acc[mt][nt], false, false);
    buf ^= 1;
  }

  // epilogue: ReLU + bf16 store
  #pragma unroll
  for (int mt = 0; mt < 2; ++mt)
    #pragma unroll
    for (int nt = 0; nt < 4; ++nt) {
      const int n = nBase + n0w + nt * 16 + lm;
      #pragma unroll
      for (int r = 0; r < 8; ++r) {
        const int m = mBase + m0w + mt * 16 + hlf * 8 + r;
        float v = acc[mt][nt][r];
        v = v > 0.0f ? v : 0.0f;
        Hout[(size_t)m * N + n] = f2bf_raw(v);
      }
    }
}

// ---------------------------------------------------------------------------
// 6) ffn2 + fused combine: out[tok] = p * (H @ w2[e]),  M=CAP K=4096 N=1024
// ---------------------------------------------------------------------------
__global__ __launch_bounds__(256) void moe_ffn2_kernel(
    const unsigned short* __restrict__ Hb,   // [E, CAP, F] bf16
    const float* __restrict__ w2,            // [E, F, D] fp32
    const int* __restrict__ slot,            // [E, CAP] token or -1
    const float* __restrict__ best_p,        // [T]
    float* __restrict__ out) {               // [T, D]
  constexpr int K = Ff, N = Dd, NK = K / 32;
  __shared__ unsigned short As[2][128][32];
  __shared__ unsigned short Bs[2][128][32];

  const int e = blockIdx.z;
  const int mBase = blockIdx.y * 128;
  const int nBase = blockIdx.x * 128;
  const unsigned short* A = Hb + (size_t)e * CAPc * K;
  const float* W = w2 + (size_t)e * K * N;

  const int tid  = threadIdx.x;
  const int lane = tid & 31;
  const int wave = tid >> 5;
  const int m0w  = (wave & 3) * 32;
  const int n0w  = (wave >> 2) * 64;
  const int hlf  = lane >> 4;
  const int lm   = lane & 15;

  v8f acc[2][4];
  const v8f vzero = {0.f,0.f,0.f,0.f,0.f,0.f,0.f,0.f};
  #pragma unroll
  for (int i = 0; i < 2; ++i)
    #pragma unroll
    for (int j = 0; j < 4; ++j) acc[i][j] = vzero;

  const int rA = tid >> 1, csA = (tid & 1) * 16;
  const int nB = tid >> 1, ksB = (tid & 1) * 16;

  auto stageA = [&](int kb, int buf) {
    const unsigned dst = (unsigned)(size_t)&As[buf][rA][csA];
    const unsigned short* src = A + (size_t)(mBase + rA) * K + kb * 32 + csA;
    async_ld_b128(dst,      src);
    async_ld_b128(dst + 16, src + 8);
  };
  auto stageB = [&](int kb, int buf) {
    const float* src = W + (size_t)(kb * 32 + ksB) * N + (nBase + nB);
    unsigned short tmp[16];
    #pragma unroll
    for (int i = 0; i < 16; ++i) tmp[i] = f2bf_raw(src[(size_t)i * N]);
    uint4* dst = (uint4*)&Bs[buf][nB][ksB];
    dst[0] = *(const uint4*)&tmp[0];
    dst[1] = *(const uint4*)&tmp[8];
  };

  stageA(0, 0);
  stageB(0, 0);

  int buf = 0;
  for (int kb = 0; kb < NK; ++kb) {
    wait_async0();
    __syncthreads();
    if (kb + 1 < NK) {
      stageA(kb + 1, buf ^ 1);
      stageB(kb + 1, buf ^ 1);
      __builtin_prefetch(W + (size_t)((kb + 1) * 32 + ksB) * N + (nBase + nB), 0, 3);
    }

    AFrag a[2], b[4];
    #pragma unroll
    for (int mt = 0; mt < 2; ++mt) {
      const int m = m0w + mt * 16 + lm;
      a[mt].q[0] = *(const uint4*)&As[buf][m][hlf * 16];
      a[mt].q[1] = *(const uint4*)&As[buf][m][hlf * 16 + 8];
    }
    #pragma unroll
    for (int nt = 0; nt < 4; ++nt) {
      const int n = n0w + nt * 16 + lm;
      b[nt].q[0] = *(const uint4*)&Bs[buf][n][hlf * 16];
      b[nt].q[1] = *(const uint4*)&Bs[buf][n][hlf * 16 + 8];
    }
    #pragma unroll
    for (int mt = 0; mt < 2; ++mt)
      #pragma unroll
      for (int nt = 0; nt < 4; ++nt)
        acc[mt][nt] = __builtin_amdgcn_wmma_f32_16x16x32_bf16(
            false, a[mt].v, false, b[nt].v, (short)0, acc[mt][nt], false, false);
    buf ^= 1;
  }

  // fused combine epilogue: slot -> token, scale by prob, scatter to d_out
  #pragma unroll
  for (int mt = 0; mt < 2; ++mt) {
    int   tok[8];
    float pp[8];
    #pragma unroll
    for (int r = 0; r < 8; ++r) {
      const int m = mBase + m0w + mt * 16 + hlf * 8 + r;
      tok[r] = slot[e * CAPc + m];
      pp[r]  = (tok[r] >= 0) ? best_p[tok[r]] : 0.0f;
    }
    #pragma unroll
    for (int nt = 0; nt < 4; ++nt) {
      const int n = nBase + n0w + nt * 16 + lm;
      #pragma unroll
      for (int r = 0; r < 8; ++r)
        if (tok[r] >= 0)
          out[(size_t)tok[r] * N + n] = pp[r] * acc[mt][nt][r];
    }
  }
}

// ---------------------------------------------------------------------------
// Launcher (graph-capture safe: stream-only, workspace carved from d_ws)
// ---------------------------------------------------------------------------
extern "C" void kernel_launch(void* const* d_in, const int* in_sizes, int n_in,
                              void* d_out, int out_size, void* d_ws, size_t ws_size,
                              hipStream_t stream) {
  const float* tokens = (const float*)d_in[0];
  const float* gate_w = (const float*)d_in[1];
  const float* w1     = (const float*)d_in[2];
  const float* w2     = (const float*)d_in[3];
  float* out = (float*)d_out;

  char* ws = (char*)d_ws;
  int*   best_e = (int*)  (ws + 0);
  float* best_p = (float*)(ws + 32768);
  int*   pos    = (int*)  (ws + 65536);
  int*   slot   = (int*)  (ws + 98304);
  unsigned short* Xe = (unsigned short*)(ws + 139264);
  unsigned short* Hb = (unsigned short*)(ws + 139264 + (size_t)Ee * CAPc * Dd * 2);

  moe_init_kernel<<<((size_t)Tt * Dd + 255) / 256, 256, 0, stream>>>(out, slot);
  moe_gate_kernel<<<Tt / 8, 256, 0, stream>>>(tokens, gate_w, best_e, best_p);
  moe_scan_kernel<<<1, 256, 0, stream>>>(best_e, pos, slot);
  moe_dispatch_kernel<<<Tt, 256, 0, stream>>>(tokens, best_e, pos, Xe);
  moe_ffn1_kernel<<<dim3(Ff / 128, CAPc / 128, Ee), 256, 0, stream>>>(Xe, w1, Hb);
  moe_ffn2_kernel<<<dim3(Dd / 128, CAPc / 128, Ee), 256, 0, stream>>>(Hb, w2, slot, best_p, out);
}
